// ExpertParallelWrapper_90305982366000
// MI455X (gfx1250) — matrix operand order repaired and checked by reading, not compiled, not run on hardware
//
#include <hip/hip_runtime.h>

#define NE      16
#define DIN     1024
#define DHID    4096
#define DOUT    1024
#define BATCH   8192
#define TILE_M  16
#define CHUNK   512
#define TILES_PER_E (BATCH / TILE_M)   // 512

#define SX_STRIDE (DIN + 8)      // ushort elems; 2064B rows (16B-aligned, 4-bank skew/row)
#define SH_STRIDE (CHUNK + 8)    // ushort elems; 1040B rows (16B-aligned, 4-bank skew/row)

typedef __attribute__((ext_vector_type(16))) __bf16    v16bf;
typedef __attribute__((ext_vector_type(8)))  float     v8f;
typedef __attribute__((ext_vector_type(8)))  unsigned  v8u;
typedef __attribute__((ext_vector_type(4)))  unsigned  v4u;

// ---------------- helpers ----------------

// Two f32 -> one u32 holding {bf16(hi), bf16(lo)} (truncation) -- single v_perm_b32.
__device__ __forceinline__ unsigned pack_bf16(float lo, float hi) {
    unsigned a = __builtin_bit_cast(unsigned, lo);
    unsigned b = __builtin_bit_cast(unsigned, hi);
#if defined(__AMDGCN__) && __has_builtin(__builtin_amdgcn_perm)
    // D.byte0=a.byte2, D.byte1=a.byte3, D.byte2=b.byte2, D.byte3=b.byte3
    return __builtin_amdgcn_perm(b, a, 0x07060302u);
#else
    return (a >> 16) | (b & 0xFFFF0000u);
#endif
}

__device__ __forceinline__ unsigned short f2bf_bits(float f) {
    return (unsigned short)(__builtin_bit_cast(unsigned, f) >> 16);   // truncate
}

// A-matrix fragment (16x32 bf16) from LDS: two 16B-aligned b128 loads.
// A layout (matches B): lanes 0-15 row M=lane, halves0..15 -> K=kb..kb+15;
// lanes 16-31 same rows with K offset +16.
__device__ __forceinline__ v16bf load_frag_A(const unsigned short* s, int stride, int kbase) {
    int lane = threadIdx.x & 31;
    int row  = lane & 15;
    int koff = kbase + ((lane & 16) ? 16 : 0);
    const v4u* p = (const v4u*)(s + row * stride + koff);
    v4u lo = p[0];        // K = koff   .. koff+7
    v4u hi = p[1];        // K = koff+8 .. koff+15
    v8u u;
#pragma unroll
    for (int i = 0; i < 4; ++i) { u[i] = lo[i]; u[4 + i] = hi[i]; }
    return __builtin_bit_cast(v16bf, u);
}

// B-matrix fragment (32x16 bf16) from global row-major f32 W[K x ldn]:
// lanes 0-15 hold K=kb+0..15 at column N=ncol0+lane; lanes 16-31 hold K=kb+16..31.
// K-pairs of f32 are packed to bf16x2 with one v_perm_b32 each.
__device__ __forceinline__ v16bf load_frag_B(const float* __restrict__ W, int ldn,
                                             int kbase, int ncol0) {
    int lane = threadIdx.x & 31;
    int n  = ncol0 + (lane & 15);
    int k0 = kbase + ((lane & 16) ? 16 : 0);
    const float* p = W + (long)k0 * ldn + n;
    v8u u;
#pragma unroll
    for (int i = 0; i < 8; ++i) {
        float lo = p[(long)(2 * i) * ldn];
        float hi = p[(long)(2 * i + 1) * ldn];
        u[i] = pack_bf16(lo, hi);
    }
    return __builtin_bit_cast(v16bf, u);
}

// ---------------- kernel 0: zero output + expert counters ----------------

__global__ void zero_kernel(float* __restrict__ out, int n, int* __restrict__ cnt) {
    long i = (long)blockIdx.x * blockDim.x + threadIdx.x;
    if (i < n)  out[i] = 0.0f;
    if (i < NE) cnt[i] = 0;
}

// ---------------- kernel 1: gating + dispatch-list build ----------------

__global__ __launch_bounds__(256) void gate_kernel(
    const float* __restrict__ x, const float* __restrict__ Wg,
    int* __restrict__ cnt, int* __restrict__ tok_list, float* __restrict__ wt_list)
{
    __shared__ float red[256][NE];   // 16 KB
    int b   = blockIdx.x;
    int tid = threadIdx.x;

    float part[NE];
#pragma unroll
    for (int e = 0; e < NE; ++e) part[e] = 0.0f;

    for (int i = tid; i < DIN; i += 256) {
        float xv = x[(long)b * DIN + i];
        const float* wr = Wg + i * NE;
#pragma unroll
        for (int e = 0; e < NE; ++e) part[e] += xv * wr[e];
    }
#pragma unroll
    for (int e = 0; e < NE; ++e) red[tid][e] = part[e];
    __syncthreads();

    for (int s = 128; s > 0; s >>= 1) {
        if (tid < s) {
#pragma unroll
            for (int e = 0; e < NE; ++e) red[tid][e] += red[tid + s][e];
        }
        __syncthreads();
    }

    if (tid == 0) {
        float v0 = -3.4e38f, v1 = -3.4e38f;
        int   i0 = 0, i1 = 0;
#pragma unroll
        for (int e = 0; e < NE; ++e) {
            float l = red[0][e];
            if (l > v0)      { v1 = v0; i1 = i0; v0 = l; i0 = e; }
            else if (l > v1) { v1 = l; i1 = e; }
        }
        float e1 = __expf(v1 - v0);      // v0 >= v1
        float inv = 1.0f / (1.0f + e1);
        float w0 = inv, w1 = e1 * inv;

        int s0 = atomicAdd(&cnt[i0], 1);
        tok_list[i0 * BATCH + s0] = b;
        wt_list[i0 * BATCH + s0]  = w0;
        int s1 = atomicAdd(&cnt[i1], 1);
        tok_list[i1 * BATCH + s1] = b;
        wt_list[i1 * BATCH + s1]  = w1;
    }
}

// ---------------- kernel 2: fused per-expert MLP (WMMA bf16) ----------------

__global__ __launch_bounds__(256) void moe_expert_kernel(
    const float* __restrict__ x,
    const float* __restrict__ W1, const float* __restrict__ b1,
    const float* __restrict__ W2, const float* __restrict__ b2,
    const int*  __restrict__ cnt,
    const int*  __restrict__ tok_list, const float* __restrict__ wt_list,
    float* __restrict__ out)
{
    __shared__ unsigned short sX[TILE_M * SX_STRIDE];  // 16 x 1024 bf16 (padded) ~33 KB
    __shared__ unsigned short sH[TILE_M * SH_STRIDE];  // 16 x 512  bf16 (padded) ~17 KB
    __shared__ int   sTok[TILE_M];
    __shared__ float sWt[TILE_M];

    int e    = blockIdx.x >> 9;              // TILES_PER_E == 512
    int tile = blockIdx.x & (TILES_PER_E - 1);
    int n_e  = cnt[e];
    if (tile * TILE_M >= n_e) return;
    int valid = n_e - tile * TILE_M;
    if (valid > TILE_M) valid = TILE_M;

    int tid = threadIdx.x;
    if (tid < TILE_M) {
        int slot = tile * TILE_M + tid;
        sTok[tid] = (tid < valid) ? tok_list[e * BATCH + slot] : 0;
        sWt[tid]  = (tid < valid) ? wt_list[e * BATCH + slot] : 0.0f;
    }
    __syncthreads();

    // Gather 16 token rows, f32 -> packed bf16x2, into LDS (coalesced pair loads)
    for (int idx = tid; idx < TILE_M * (DIN / 2); idx += 256) {
        int row  = idx >> 9;               // 512 pairs per row
        int col2 = (idx & 511) * 2;
        float lo = 0.0f, hi = 0.0f;
        if (row < valid) {
            const float* xp = x + (long)sTok[row] * DIN + col2;
            lo = xp[0]; hi = xp[1];
        }
        *(unsigned*)(sX + row * SX_STRIDE + col2) = pack_bf16(lo, hi);
    }
    __syncthreads();

    int wave = tid >> 5;
    int lane = tid & 31;
    int ccol = lane & 15;                 // C/D fragment column
    int rsel = (lane & 16) ? 8 : 0;       // C/D fragment row offset

    const float* W1e = W1 + (long)e * DIN * DHID;
    const float* W2e = W2 + (long)e * DHID * DOUT;

    // Persistent Y accumulators: wave owns 8 N-tiles (128 output columns)
    v8f yacc[8];
#pragma unroll
    for (int j = 0; j < 8; ++j)
#pragma unroll
        for (int r = 0; r < 8; ++r) yacc[j][r] = 0.0f;

    for (int c = 0; c < DHID / CHUNK; ++c) {
        // ---- phase A: Hc = relu(X @ W1[:, c*CHUNK : (c+1)*CHUNK] + b1) ----
        v8f hacc[4];
#pragma unroll
        for (int j = 0; j < 4; ++j)
#pragma unroll
            for (int r = 0; r < 8; ++r) hacc[j][r] = 0.0f;

        for (int kb = 0; kb < DIN; kb += 32) {
            // branch-free prefetch of the next K-slab (clamped to stay in-bounds)
            int kpf = (kb + 32 < DIN) ? kb + 32 : kb;
            __builtin_prefetch(W1e + (long)kpf * DHID + c * CHUNK + wave * 64, 0, 0);
            v16bf a = load_frag_A(sX, SX_STRIDE, kb);
#pragma unroll
            for (int j = 0; j < 4; ++j) {
                int hcol0 = (wave * 4 + j) * 16;     // chunk-local column tile
                v16bf b = load_frag_B(W1e, DHID, kb, c * CHUNK + hcol0);
                hacc[j] = __builtin_amdgcn_wmma_f32_16x16x32_bf16(
                    false, a, false, b, (short)0, hacc[j], false, false);
            }
        }
        // bias + relu + bf16 store to sH
#pragma unroll
        for (int j = 0; j < 4; ++j) {
            int hcol0 = (wave * 4 + j) * 16;
            int gcol  = c * CHUNK + hcol0 + ccol;
            float bias = b1[e * DHID + gcol];
#pragma unroll
            for (int r = 0; r < 8; ++r) {
                float v = hacc[j][r] + bias;
                v = v > 0.0f ? v : 0.0f;
                sH[(r + rsel) * SH_STRIDE + hcol0 + ccol] = f2bf_bits(v);
            }
        }
        __syncthreads();

        // ---- phase B: Y += Hc @ W2[c*CHUNK : (c+1)*CHUNK, :] ----
        const float* W2c = W2e + (long)c * CHUNK * DOUT;
        for (int kb = 0; kb < CHUNK; kb += 32) {
            int kpf = (kb + 32 < CHUNK) ? kb + 32 : kb;
            __builtin_prefetch(W2c + (long)kpf * DOUT + wave * 128, 0, 0);
            v16bf a = load_frag_A(sH, SH_STRIDE, kb);
#pragma unroll
            for (int j = 0; j < 8; ++j) {
                int ncol0 = (wave * 8 + j) * 16;
                v16bf b = load_frag_B(W2c, DOUT, kb, ncol0);
                yacc[j] = __builtin_amdgcn_wmma_f32_16x16x32_bf16(
                    false, a, false, b, (short)0, yacc[j], false, false);
            }
        }
        __syncthreads();   // before next chunk overwrites sH
    }

    // ---- epilogue: out[token] += w * (y + b2), f32 global atomics ----
#pragma unroll
    for (int j = 0; j < 8; ++j) {
        int col = (wave * 8 + j) * 16 + ccol;
        float bias = b2[e * DOUT + col];
#pragma unroll
        for (int r = 0; r < 8; ++r) {
            int row = r + rsel;
            if (row < valid) {
                float v = yacc[j][r] + bias;
                atomicAdd(&out[(long)sTok[row] * DOUT + col], sWt[row] * v);
            }
        }
    }
}

// ---------------- launcher ----------------

extern "C" void kernel_launch(void* const* d_in, const int* in_sizes, int n_in,
                              void* d_out, int out_size, void* d_ws, size_t ws_size,
                              hipStream_t stream) {
    const float* x  = (const float*)d_in[0];
    const float* Wg = (const float*)d_in[1];
    const float* W1 = (const float*)d_in[2];
    const float* b1 = (const float*)d_in[3];
    const float* W2 = (const float*)d_in[4];
    const float* b2 = (const float*)d_in[5];
    float* out = (float*)d_out;

    char*  ws  = (char*)d_ws;
    int*   cnt = (int*)ws;                                      // 16 counters (256B slot)
    int*   tok = (int*)(ws + 256);                              // 16 x 8192 int
    float* wt  = (float*)(ws + 256 + (size_t)NE * BATCH * 4);   // 16 x 8192 float

    int n = BATCH * DOUT;
    zero_kernel<<<(n + 255) / 256, 256, 0, stream>>>(out, n, cnt);
    gate_kernel<<<BATCH, 256, 0, stream>>>(x, Wg, cnt, tok, wt);
    moe_expert_kernel<<<NE * TILES_PER_E, 256, 0, stream>>>(x, W1, b1, W2, b2,
                                                            cnt, tok, wt, out);
}
